// MPNN_69672959475921
// MI455X (gfx1250) — matrix operand order repaired and checked by reading, not compiled, not run on hardware
//
#include <hip/hip_runtime.h>
#include <hip/hip_bf16.h>

#define NNODES 20000
#define NEDGES 160000
#define NGRP   64
#define COUTN  10
#define EPSV   1e-5f

typedef __attribute__((ext_vector_type(16))) __bf16 v16bf;
typedef __attribute__((ext_vector_type(8)))  float  v8f;

union BFrag {
  v16bf v;
  unsigned short u[16];
  unsigned int   w[8];
  uint4 q[2];
};

__device__ __forceinline__ unsigned int rne_bf(float f) {
  unsigned int x = __float_as_uint(f);
  return x + (0x7FFFu + ((x >> 16) & 1u));   // bf16 RNE in bits [31:16]
}
// pack2(a,b): low 16 = bf16(a), high 16 = bf16(b)  (one v_perm_b32 for the pack)
__device__ __forceinline__ unsigned int pack2(float a, float b) {
  return __builtin_amdgcn_perm(rne_bf(b), rne_bf(a), 0x07060302u);
}
__device__ __forceinline__ unsigned short f2bf(float f) {
  return (unsigned short)(rne_bf(f) >> 16);
}
// native packed bf16 multiply (V_PK_MUL_BF16, VOP3P op 42) — no clang builtin
__device__ __forceinline__ unsigned int pkmul_bf16(unsigned int a, unsigned int b) {
  unsigned int d;
  asm("v_pk_mul_bf16 %0, %1, %2" : "=v"(d) : "v"(a), "v"(b));
  return d;
}

// ---------------------------------------------------------------------------
// Reorder edge-MLP second layer (w2 [32, cin*32], b2 [cin*32]) into the exact
// per-lane WMMA B-fragment layout, with b2 folded in as K rows [32*cin,
// 32*cin+cin) (the "h[32]=1" trick) and zero padding to a multiple of 32.
// Bs[((chunk*2+tile)*32 + lane)*16 + j] = B[chunk*32 + 16*(j/8) + 8*(lane/16) + j%8][tile*16 + lane%16]
// ---------------------------------------------------------------------------
__global__ void prep_B_kernel(const float* __restrict__ w2, const float* __restrict__ b2,
                              unsigned short* __restrict__ Bs, int cin, int nchunk) {
  int idx = blockIdx.x * blockDim.x + threadIdx.x;
  if (idx >= nchunk * 1024) return;
  int j     = idx & 15;
  int laneB = (idx >> 4) & 31;
  int tile  = (idx >> 9) & 1;
  int chunk = idx >> 10;
  int k = chunk * 32 + ((j >> 3) << 4) + ((laneB >> 4) << 3) + (j & 7);
  int o = tile * 16 + (laneB & 15);
  int kmain = 32 * cin;
  float v = 0.0f;
  if (k < kmain)              v = w2[(k / cin) * (cin * 32) + (k % cin) * 32 + o];
  else if (k < kmain + cin)   v = b2[(k - kmain) * 32 + o];
  Bs[idx] = f2bf(v);
}

// root [cin, 32] -> single-chunk B fragments (K padded to 32 with zeros)
__global__ void prep_root_kernel(const float* __restrict__ root,
                                 unsigned short* __restrict__ Rs, int cin) {
  int idx = blockIdx.x * blockDim.x + threadIdx.x;
  if (idx >= 1024) return;
  int j     = idx & 15;
  int laneB = (idx >> 4) & 31;
  int tile  = idx >> 9;
  int k = ((j >> 3) << 4) + ((laneB >> 4) << 3) + (j & 7);
  int o = tile * 16 + (laneB & 15);
  float v = (k < cin) ? root[k * 32 + o] : 0.0f;
  Rs[idx] = f2bf(v);
}

// ---------------------------------------------------------------------------
// Per-edge fused GEMM:  MSG[16e,32] = (h ⊗ x_src)[16e, 33*CIN] @ B, scattered
// with atomics into agg[dst]. One wave per 16-edge tile, 4 waves per block.
// A chunks are built with packed-bf16 multiplies (16 v_pk_mul_bf16 per chunk).
// ---------------------------------------------------------------------------
template <int CIN, bool BN>
__global__ __launch_bounds__(128)
void msg_kernel(const float* __restrict__ xin,
                const float* __restrict__ scv, const float* __restrict__ shv,
                const int* __restrict__ ei, const float* __restrict__ ewp,
                const float* __restrict__ w1, const float* __restrict__ b1,
                const unsigned short* __restrict__ Bs,
                float* __restrict__ agg) {
  constexpr int KEXT = 33 * CIN;
  constexpr int NCH  = (KEXT + 31) / 32;

  __shared__ uint4 sA4[4][16][4];   // per-wave A-chunk staging (16 rows x 32 bf16)
  __shared__ int   sdst[4][16];

  const int wv   = threadIdx.x >> 5;
  const int lane = threadIdx.x & 31;
  const int mrow = lane & 15;
  const int hx   = lane >> 4;
  const int e0   = (blockIdx.x * 4 + wv) * 16;
  const int e    = e0 + mrow;
  const bool ein = (e < NEDGES);

  float ww = ein ? ewp[e] : 0.0f;
  int  src = ein ? ei[e] : 0;
  int  dst = ein ? ei[NEDGES + e] : 0;

  // h_k = relu(ew*w1_k + b1_k), splatted into a packed bf16 pair
  unsigned int hsplat[34];
#pragma unroll
  for (int k = 0; k < 32; ++k) {
    float hv = fmaxf(fmaf(ww, w1[k], b1[k]), 0.0f);
    hsplat[k] = pack2(hv, hv);
  }
  hsplat[32] = 0x3F803F80u;  // bf16(1.0) pair: folds b2 into the GEMM
  hsplat[33] = 0u;           // zero padding rows (CIN=16 case)

  // x_src as packed bf16 pairs (optionally through fused BN+ReLU)
  unsigned int xp[CIN / 2];
#pragma unroll
  for (int i = 0; i < CIN; i += 2) {
    float v0 = xin[src * CIN + i];
    float v1 = xin[src * CIN + i + 1];
    if (BN) {
      v0 = fmaxf(fmaf(v0, scv[i],     shv[i]),     0.0f);
      v1 = fmaxf(fmaf(v1, scv[i + 1], shv[i + 1]), 0.0f);
    }
    xp[i / 2] = pack2(v0, v1);
  }
  if (lane < 16) sdst[wv][mrow] = dst;

  __builtin_prefetch(Bs, 0, 3);

  v8f acc0 = {0.f, 0.f, 0.f, 0.f, 0.f, 0.f, 0.f, 0.f};
  v8f acc1 = {0.f, 0.f, 0.f, 0.f, 0.f, 0.f, 0.f, 0.f};

// pair P (P=0..15) of chunk c covers K = c*32 + 2P, 2P+1
#define HP(P) hsplat[(c * 32 + 2 * (P)) / CIN]
#define XPX(P) xp[(((c * 32 + 2 * (P)) % CIN) >> 1)]
#pragma unroll
  for (int c = 0; c < NCH; ++c) {
    if (lane < 16) {                      // row owners build A chunk
#pragma unroll
      for (int s = 0; s < 4; ++s) {
        uint4 pk;
        pk.x = pkmul_bf16(HP(s * 4 + 0), XPX(s * 4 + 0));
        pk.y = pkmul_bf16(HP(s * 4 + 1), XPX(s * 4 + 1));
        pk.z = pkmul_bf16(HP(s * 4 + 2), XPX(s * 4 + 2));
        pk.w = pkmul_bf16(HP(s * 4 + 3), XPX(s * 4 + 3));
        sA4[wv][mrow][s] = pk;
      }
    }
    __syncthreads();
    BFrag a;
    a.q[0] = sA4[wv][mrow][hx];           // K = c*32 + 8*half + [0..7]
    a.q[1] = sA4[wv][mrow][2 + hx];       // K = c*32 + 16 + 8*half + [0..7]

    const uint4* bp = (const uint4*)Bs;
    const int base0 = ((c * 2 + 0) * 32 + lane) * 2;
    const int base1 = ((c * 2 + 1) * 32 + lane) * 2;
    BFrag b0, b1f;
    b0.q[0]  = bp[base0];     b0.q[1]  = bp[base0 + 1];
    b1f.q[0] = bp[base1];     b1f.q[1] = bp[base1 + 1];

    acc0 = __builtin_amdgcn_wmma_f32_16x16x32_bf16(false, a.v, false, b0.v,  (short)0, acc0, false, false);
    acc1 = __builtin_amdgcn_wmma_f32_16x16x32_bf16(false, a.v, false, b1f.v, (short)0, acc1, false, false);
    __syncthreads();
  }
#undef HP
#undef XPX

#pragma unroll
  for (int r = 0; r < 8; ++r) {
    int ee = e0 + r + 8 * hx;             // D row -> edge
    if (ee < NEDGES) {
      int d = sdst[wv][r + 8 * hx];
      atomicAdd(&agg[d * 32 + (lane & 15)],      acc0[r]);
      atomicAdd(&agg[d * 32 + 16 + (lane & 15)], acc1[r]);
    }
  }
}

// ---------------------------------------------------------------------------
// Node update: y = xeff @ root + agg + bias (one WMMA chunk, C preloaded with
// agg+bias), plus BN sum/sumsq accumulation. xeff optionally = relu(bn(prev)).
// ---------------------------------------------------------------------------
template <int CIN, bool BN>
__global__ __launch_bounds__(128)
void node_kernel(const float* __restrict__ xin,
                 const float* __restrict__ scv, const float* __restrict__ shv,
                 const unsigned short* __restrict__ Rs,
                 const float* __restrict__ aggp, const float* __restrict__ biasp,
                 float* __restrict__ yout, float* __restrict__ sumP, float* __restrict__ sqP) {
  const int wv   = threadIdx.x >> 5;
  const int lane = threadIdx.x & 31;
  const int mrow = lane & 15;
  const int hx   = lane >> 4;
  const int tile = blockIdx.x * 4 + wv;
  if (tile >= (NNODES + 15) / 16) return;
  const int n0   = tile * 16;
  const int node = n0 + mrow;
  const bool nin = (node < NNODES);

  // xeff as 16 packed bf16 pairs (zero padded past CIN)
  unsigned int xpair[16];
#pragma unroll
  for (int i = 0; i < 16; ++i) {
    float v0 = 0.0f, v1 = 0.0f;
    if (2 * i < CIN && nin) {
      v0 = xin[node * CIN + 2 * i];
      v1 = xin[node * CIN + 2 * i + 1];
      if (BN) {
        v0 = fmaxf(fmaf(v0, scv[2 * i],     shv[2 * i]),     0.0f);
        v1 = fmaxf(fmaf(v1, scv[2 * i + 1], shv[2 * i + 1]), 0.0f);
      }
    }
    xpair[i] = pack2(v0, v1);
  }
  BFrag a;
#pragma unroll
  for (int t = 0; t < 4; ++t) {
    a.w[t]     = hx ? xpair[4 + t]  : xpair[t];       // K = 8*half + [0..7]
    a.w[4 + t] = hx ? xpair[12 + t] : xpair[8 + t];   // K = 16 + 8*half + [0..7]
  }
  const uint4* rp = (const uint4*)Rs;
  BFrag b0, b1f;
  b0.q[0]  = rp[lane * 2];          b0.q[1]  = rp[lane * 2 + 1];
  b1f.q[0] = rp[(32 + lane) * 2];   b1f.q[1] = rp[(32 + lane) * 2 + 1];

  v8f c0, c1;
  const int col = lane & 15;
#pragma unroll
  for (int r = 0; r < 8; ++r) {
    int row = n0 + r + 8 * hx;
    float a0 = 0.0f, a1 = 0.0f;
    if (row < NNODES) {
      a0 = aggp[row * 32 + col]      + biasp[col];
      a1 = aggp[row * 32 + 16 + col] + biasp[16 + col];
    }
    c0[r] = a0; c1[r] = a1;
  }
  c0 = __builtin_amdgcn_wmma_f32_16x16x32_bf16(false, a.v, false, b0.v,  (short)0, c0, false, false);
  c1 = __builtin_amdgcn_wmma_f32_16x16x32_bf16(false, a.v, false, b1f.v, (short)0, c1, false, false);

  float s0 = 0.f, q0 = 0.f, s1 = 0.f, q1 = 0.f;
#pragma unroll
  for (int r = 0; r < 8; ++r) {
    int row = n0 + r + 8 * hx;
    if (row < NNODES) {
      float v0 = c0[r], v1 = c1[r];
      yout[row * 32 + col]      = v0;
      yout[row * 32 + 16 + col] = v1;
      s0 += v0; q0 += v0 * v0;
      s1 += v1; q1 += v1 * v1;
    }
  }
  atomicAdd(&sumP[col], s0);       atomicAdd(&sqP[col], q0);
  atomicAdd(&sumP[16 + col], s1);  atomicAdd(&sqP[16 + col], q1);
}

__global__ void finalize_bn_kernel(const float* __restrict__ sum, const float* __restrict__ sq,
                                   const float* __restrict__ gamma, const float* __restrict__ beta,
                                   float* __restrict__ sc, float* __restrict__ sh) {
  int o = threadIdx.x;
  if (o >= 32) return;
  float mean = sum[o] / (float)NNODES;
  float var  = sq[o] / (float)NNODES - mean * mean;   // biased var (jnp.var)
  float s    = gamma[o] * rsqrtf(var + EPSV);
  sc[o] = s;
  sh[o] = beta[o] - mean * s;
}

__global__ void pool_kernel(const float* __restrict__ y2, const float* __restrict__ sc,
                            const float* __restrict__ sh, const int* __restrict__ batch,
                            float* __restrict__ psum, float* __restrict__ pcnt) {
  int idx = blockIdx.x * blockDim.x + threadIdx.x;
  if (idx >= NNODES * 32) return;
  int n = idx >> 5, o = idx & 31;
  float v = fmaxf(fmaf(y2[idx], sc[o], sh[o]), 0.0f);
  int g = batch[n];
  atomicAdd(&psum[g * 32 + o], v);
  if (o == 0) atomicAdd(&pcnt[g], 1.0f);
}

__global__ void head_kernel(const float* __restrict__ psum, const float* __restrict__ pcnt,
                            const float* __restrict__ fc1w, const float* __restrict__ fc1b,
                            const float* __restrict__ fc2w, const float* __restrict__ fc2b,
                            float* __restrict__ out) {
  int g = threadIdx.x;
  if (g >= NGRP) return;
  float inv = 1.0f / fmaxf(pcnt[g], 1.0f);
  float p[32], t[32];
#pragma unroll
  for (int i = 0; i < 32; ++i) p[i] = psum[g * 32 + i] * inv;
#pragma unroll
  for (int j = 0; j < 32; ++j) {
    float acc = fc1b[j];
#pragma unroll
    for (int i = 0; i < 32; ++i) acc = fmaf(p[i], fc1w[i * 32 + j], acc);
    t[j] = fmaxf(acc, 0.0f);
  }
#pragma unroll
  for (int o = 0; o < COUTN; ++o) {
    float acc = fc2b[o];
#pragma unroll
    for (int j = 0; j < 32; ++j) acc = fmaf(t[j], fc2w[j * COUTN + o], acc);
    out[g * COUTN + o] = acc;
  }
}

extern "C" void kernel_launch(void* const* d_in, const int* in_sizes, int n_in,
                              void* d_out, int out_size, void* d_ws, size_t ws_size,
                              hipStream_t stream) {
  (void)in_sizes; (void)n_in; (void)out_size; (void)ws_size;
  const float* x     = (const float*)d_in[0];
  const int*   ei    = (const int*)d_in[1];
  const int*   batch = (const int*)d_in[2];
  const float* ew    = (const float*)d_in[3];
  const float* e1w1  = (const float*)d_in[4];
  const float* e1b1  = (const float*)d_in[5];
  const float* e1w2  = (const float*)d_in[6];
  const float* e1b2  = (const float*)d_in[7];
  const float* root1 = (const float*)d_in[8];
  const float* bias1 = (const float*)d_in[9];
  const float* e2w1  = (const float*)d_in[10];
  const float* e2b1  = (const float*)d_in[11];
  const float* e2w2  = (const float*)d_in[12];
  const float* e2b2  = (const float*)d_in[13];
  const float* root2 = (const float*)d_in[14];
  const float* bias2 = (const float*)d_in[15];
  const float* gamma = (const float*)d_in[16];
  const float* beta  = (const float*)d_in[17];
  const float* fc1w  = (const float*)d_in[18];
  const float* fc1b  = (const float*)d_in[19];
  const float* fc2w  = (const float*)d_in[20];
  const float* fc2b  = (const float*)d_in[21];
  float* out = (float*)d_out;

  const int NCH1 = (33 * 16 + 31) / 32;  // 17 chunks (K=528 padded to 544)
  const int NCH2 = (33 * 32) / 32;       // 33 chunks (K=1056)

  char* base = (char*)d_ws;
  size_t off = 0;
  auto carve = [&](size_t bytes) -> char* {
    char* p = base + off;
    off = (off + bytes + 255) & ~(size_t)255;
    return p;
  };
  unsigned short* Bs1 = (unsigned short*)carve((size_t)NCH1 * 1024 * 2);
  unsigned short* Bs2 = (unsigned short*)carve((size_t)NCH2 * 1024 * 2);
  unsigned short* Rs1 = (unsigned short*)carve(1024 * 2);
  unsigned short* Rs2 = (unsigned short*)carve(1024 * 2);
  float* agg1 = (float*)carve((size_t)NNODES * 32 * 4);
  float* y1   = (float*)carve((size_t)NNODES * 32 * 4);
  float* agg2 = (float*)carve((size_t)NNODES * 32 * 4);
  float* y2   = (float*)carve((size_t)NNODES * 32 * 4);
  float* sum1 = (float*)carve(32 * 4);
  float* sq1  = (float*)carve(32 * 4);
  float* sum2 = (float*)carve(32 * 4);
  float* sq2  = (float*)carve(32 * 4);
  float* sc1  = (float*)carve(32 * 4);
  float* sh1  = (float*)carve(32 * 4);
  float* sc2  = (float*)carve(32 * 4);
  float* sh2  = (float*)carve(32 * 4);
  float* psum = (float*)carve((size_t)NGRP * 32 * 4);
  float* pcnt = (float*)carve((size_t)NGRP * 4);

  hipMemsetAsync(agg1, 0, (size_t)NNODES * 32 * 4, stream);
  hipMemsetAsync(agg2, 0, (size_t)NNODES * 32 * 4, stream);
  hipMemsetAsync(sum1, 0, 32 * 4, stream);
  hipMemsetAsync(sq1,  0, 32 * 4, stream);
  hipMemsetAsync(sum2, 0, 32 * 4, stream);
  hipMemsetAsync(sq2,  0, 32 * 4, stream);
  hipMemsetAsync(psum, 0, (size_t)NGRP * 32 * 4, stream);
  hipMemsetAsync(pcnt, 0, (size_t)NGRP * 4, stream);

  prep_B_kernel<<<(NCH1 * 1024 + 255) / 256, 256, 0, stream>>>(e1w2, e1b2, Bs1, 16, NCH1);
  prep_B_kernel<<<(NCH2 * 1024 + 255) / 256, 256, 0, stream>>>(e2w2, e2b2, Bs2, 32, NCH2);
  prep_root_kernel<<<4, 256, 0, stream>>>(root1, Rs1, 16);
  prep_root_kernel<<<4, 256, 0, stream>>>(root2, Rs2, 32);

  const int eblocks = ((NEDGES + 15) / 16 + 3) / 4;   // 2500
  const int nblocks = ((NNODES + 15) / 16 + 3) / 4;   // 313

  // Layer 1
  msg_kernel<16, false><<<eblocks, 128, 0, stream>>>(x, nullptr, nullptr, ei, ew, e1w1, e1b1, Bs1, agg1);
  node_kernel<16, false><<<nblocks, 128, 0, stream>>>(x, nullptr, nullptr, Rs1, agg1, bias1, y1, sum1, sq1);
  finalize_bn_kernel<<<1, 32, 0, stream>>>(sum1, sq1, gamma, beta, sc1, sh1);

  // Layer 2 (BN+ReLU of layer-1 output fused into the loads)
  msg_kernel<32, true><<<eblocks, 128, 0, stream>>>(y1, sc1, sh1, ei, ew, e2w1, e2b1, Bs2, agg2);
  node_kernel<32, true><<<nblocks, 128, 0, stream>>>(y1, sc1, sh1, Rs2, agg2, bias2, y2, sum2, sq2);
  finalize_bn_kernel<<<1, 32, 0, stream>>>(sum2, sq2, gamma + 32, beta + 32, sc2, sh2);

  // Mean-pool + MLP head
  pool_kernel<<<(NNODES * 32 + 255) / 256, 256, 0, stream>>>(y2, sc2, sh2, batch, psum, pcnt);
  head_kernel<<<1, 64, 0, stream>>>(psum, pcnt, fc1w, fc1b, fc2w, fc2b, out);
}
